// FavorPlusAttention_84524956386129
// MI455X (gfx1250) — compile-verified, matrix operand order checked by reading, not run on hardware
//
#include <hip/hip_runtime.h>
#include <hip/hip_bf16.h>
#include <math.h>

// ---------------------------------------------------------------------------
// FAVOR+ (Performer) attention for MI455X / gfx1250, wave32 + WMMA bf16.
// GEMMs use v_wmma_f32_16x16x32_bf16 with LDS-staged fragments (ISA 7.12.2).
// 2D tile loads go through the Tensor Data Mover (tensor_load_to_lds, 6-arg
// clang-23 builtin) where available; manual staging is the fallback.
// ---------------------------------------------------------------------------

typedef __bf16 bf16_t;
typedef __attribute__((ext_vector_type(16))) __bf16 v16bf;
typedef __attribute__((ext_vector_type(8)))  __bf16 v8bf;
typedef __attribute__((ext_vector_type(8)))  float  v8f;

#define B_   4
#define N_   4096
#define DM   1024
#define NH   16
#define DH   64
#define BH   (B_*NH)     // 64 (b,h) pairs
#define MF   266         // nb_features
#define MP   288         // padded features = 18 * 16
#define EPSF 1e-4f

__device__ __forceinline__ bf16_t f2b(float f){ return (bf16_t)f; }
__device__ __forceinline__ float  b2f(bf16_t b){ return (float)b; }

__device__ __forceinline__ v8f wmma_bf16(v16bf a, v16bf b, v8f c){
  return __builtin_amdgcn_wmma_f32_16x16x32_bf16(false, a, false, b,
                                                 (short)0, c, false, false);
}

// Load a 16x32 (rows x K) bf16 fragment from row-major LDS (stride in elems).
// ISA 7.12.2: lane holds row (lane&15); half=lane>>4 selects K 8-groups.
__device__ __forceinline__ v16bf frag16(const bf16_t* p, int stride){
  const int lane = threadIdx.x & 31;
  const bf16_t* r = p + (lane & 15) * stride + ((lane >> 4) << 3);
  v16bf f;
#pragma unroll
  for (int i = 0; i < 8; ++i){ f[i] = r[i]; f[8 + i] = r[16 + i]; }
  return f;
}

// --------------------------- TDM tile loader -------------------------------
#if defined(__has_builtin)
#if __has_builtin(__builtin_amdgcn_tensor_load_to_lds) && \
    __has_builtin(__builtin_amdgcn_s_wait_tensorcnt)
#define HAVE_TDM 1
#endif
#endif

#ifdef HAVE_TDM
typedef __attribute__((ext_vector_type(4))) unsigned int tdm_g0_t;
typedef __attribute__((ext_vector_type(8))) int          tdm_g1_t;
typedef __attribute__((ext_vector_type(4))) int          tdm_g2_t;

// Load a 2D tile (tile_w x tile_h, 2-byte elems, row stride in elems) from
// global into LDS at lds_addr (bytes). D# layout per CDNA5 ISA ch.8.
__device__ __forceinline__ void tdm_load_tile_2d(unsigned lds_addr,
                                                 const void* gptr,
                                                 unsigned tile_w,
                                                 unsigned tile_h,
                                                 unsigned long long stride_el,
                                                 unsigned tensor_w){
  const unsigned long long ga = (unsigned long long)gptr;
  const unsigned td1 = 1u << 20;              // tensor_dim1 (OOB bound)
  tdm_g0_t g0;
  g0[0] = 1u;                                  // count=1, user descriptor
  g0[1] = lds_addr;                            // LDS byte address
  g0[2] = (unsigned)ga;                        // global_addr[31:0]
  g0[3] = (unsigned)((ga >> 32) & 0x01FFFFFFu) // global_addr[56:32]
        | (2u << 30);                          // type=2 ("image")
  tdm_g1_t g1;
  g1[0] = 0x00010000;                          // data_size=1 -> 2 bytes
  g1[1] = (int)((tensor_w & 0xFFFFu) << 16);   // tensor_dim0[15:0] @bit48
  g1[2] = (int)((tensor_w >> 16) | ((td1 & 0xFFFFu) << 16));
  g1[3] = (int)((td1 >> 16) | (tile_w << 16)); // tile_dim0 @bit112
  g1[4] = (int)(tile_h & 0xFFFFu);             // tile_dim1 (tile_dim2=0)
  g1[5] = (int)(unsigned)stride_el;            // tensor_dim0_stride[31:0]
  g1[6] = (int)((stride_el >> 32) & 0xFFFFu);  // stride[47:32] | dim1_stride lo
  g1[7] = 0;
  tdm_g2_t z4; z4[0]=0; z4[1]=0; z4[2]=0; z4[3]=0;
  tdm_g1_t z8;
#pragma unroll
  for (int i = 0; i < 8; ++i) z8[i] = 0;
  // 6-arg form (clang-23 / therock-10.0): groups 2,3 unused for 2D tensors.
  __builtin_amdgcn_tensor_load_to_lds(g0, g1, z4, z4, z8, 0);
}
__device__ __forceinline__ unsigned lds_off(const void* p){
  return (unsigned)(size_t)p;                  // low 32b of shared flat addr
}
#endif

// ------------------------------ conversions --------------------------------

__global__ void cvt_bf16_kernel(const float* __restrict__ in,
                                bf16_t* __restrict__ out, int n){
  int i = blockIdx.x * blockDim.x + threadIdx.x;
  if (i < n) out[i] = f2b(in[i]);
}

// WT[n][k] = W[k][n]
__global__ void transpose_cvt_1024(const float* __restrict__ W,
                                   bf16_t* __restrict__ WT){
  int i = blockIdx.x * blockDim.x + threadIdx.x;
  int nn = i >> 10, k = i & 1023;
  WT[i] = f2b(W[k * DM + nn]);
}

__global__ void proj_pad_kernel(const float* __restrict__ proj,
                                bf16_t* __restrict__ out){
  int i = blockIdx.x * blockDim.x + threadIdx.x;
  if (i >= MP * DH) return;
  int m = i / DH, d = i % DH;
  out[i] = (m < MF) ? f2b(proj[m * DH + d]) : (bf16_t)0.0f;
}

// --------------------------- big 1024-K GEMM -------------------------------
// Block tile 128x64, 4 waves, each wave 32x64 (8 accumulators).
template<bool AGATHER, bool OUTHEADS>
__global__ __launch_bounds__(128)
void gemm1024_kernel(const bf16_t* __restrict__ A, const bf16_t* __restrict__ BT,
                     const float* __restrict__ bias,
                     bf16_t* __restrict__ outB, float* __restrict__ outF){
  __shared__ bf16_t ldsA[128 * 32];
  __shared__ bf16_t ldsB[64 * 32];
  const int rowBase = blockIdx.y * 128;
  const int colBase = blockIdx.x * 64;
  const int t = threadIdx.x;
  const int wv = t >> 5, lane = t & 31, half = lane >> 4, ln = lane & 15;

  v8f acc[2][4];
#pragma unroll
  for (int rt = 0; rt < 2; ++rt)
#pragma unroll
    for (int c = 0; c < 4; ++c)
#pragma unroll
      for (int i = 0; i < 8; ++i) acc[rt][c][i] = 0.f;

  for (int k0 = 0; k0 < DM; k0 += 32){
    __syncthreads();
    // stage A tile 128x32 (manual, coalesced 16B chunks)
#pragma unroll
    for (int i = 0; i < 4; ++i){
      int chunk = t + 128 * i;
      int r = chunk >> 2, cc = (chunk & 3) << 3;
      int gr = rowBase + r;
      const bf16_t* src;
      if (AGATHER){
        int b = gr >> 12, n = gr & 4095;
        int kk = k0 + cc;
        int h = kk >> 6, d = kk & 63;
        src = A + (((size_t)(b * NH + h) * N_ + n) * DH + d);
      } else {
        src = A + ((size_t)gr * DM + k0 + cc);
      }
      if (k0 + 32 < DM) __builtin_prefetch(src + 32, 0, 3); // global_prefetch_b8
      *(v8bf*)(ldsA + r * 32 + cc) = *(const v8bf*)src;
    }
    // stage B^T tile 64x32
#ifdef HAVE_TDM
    if (wv == 0){
      tdm_load_tile_2d(lds_off(ldsB), BT + ((size_t)colBase * DM + k0),
                       32, 64, DM, DM);
      __builtin_amdgcn_s_wait_tensorcnt(0);
    }
#else
#pragma unroll
    for (int i = 0; i < 2; ++i){
      int chunk = t + 128 * i;
      int r = chunk >> 2, cc = (chunk & 3) << 3;
      *(v8bf*)(ldsB + r * 32 + cc) =
          *(const v8bf*)(BT + ((size_t)(colBase + r) * DM + k0 + cc));
    }
#endif
    __syncthreads();
    v16bf va0 = frag16(ldsA + (wv * 32) * 32, 32);
    v16bf va1 = frag16(ldsA + (wv * 32 + 16) * 32, 32);
#pragma unroll
    for (int c = 0; c < 4; ++c){
      v16bf vb = frag16(ldsB + c * 16 * 32, 32);
      acc[0][c] = wmma_bf16(va0, vb, acc[0][c]);
      acc[1][c] = wmma_bf16(va1, vb, acc[1][c]);
    }
  }

#pragma unroll
  for (int rt = 0; rt < 2; ++rt)
#pragma unroll
    for (int c = 0; c < 4; ++c)
#pragma unroll
      for (int r = 0; r < 8; ++r){
        int gm = rowBase + wv * 32 + rt * 16 + r + 8 * half;
        int gc = colBase + c * 16 + ln;
        float v = acc[rt][c][r];
        if (OUTHEADS){
          int b = gm >> 12, n = gm & 4095;
          int h = gc >> 6, d = gc & 63;
          outB[((size_t)(b * NH + h) * N_ + n) * DH + d] = f2b(v);
        } else {
          outF[(size_t)gm * DM + gc] = v + bias[gc];
        }
      }
}

// ------------------------- feature map (FAVOR+) ----------------------------
enum { FEAT_Q = 0, FEAT_K_STATS = 1, FEAT_K_FINAL = 2 };

template<int MODE>
__global__ __launch_bounds__(128)
void feat_kernel(const bf16_t* __restrict__ qk,     // [BH][N][64]
                 const bf16_t* __restrict__ projb,  // [MP][64]
                 bf16_t* __restrict__ outQP,        // Q: [BH][N][MP]
                 bf16_t* __restrict__ outKPT,       // K_FINAL: [BH][MP][N]
                 float* __restrict__ rowmax_out,    // K_STATS: [BH][N]
                 const float* __restrict__ stab_in){// K_FINAL: [BH]
  __shared__ bf16_t ldsA[64 * 64];
  __shared__ bf16_t ldsP[MP * 64];
  __shared__ float  diagBuf[4][16];

  const int bh = blockIdx.y;
  const int rowBase = blockIdx.x * 64;
  const int t = threadIdx.x;
  const int wv = t >> 5, lane = t & 31, half = lane >> 4, ln = lane & 15;
  const float dn = 0.35355339059327373f;       // 64^-0.25
  const float ratio = rsqrtf((float)MF);

  // stage projection table [MP][64] once
#ifdef HAVE_TDM
  if (wv == 0){
    tdm_load_tile_2d(lds_off(ldsP), projb, 64, MP, 64, 64);
    __builtin_amdgcn_s_wait_tensorcnt(0);
  }
#else
  for (int ch = t; ch < (MP * 64) / 8; ch += 128){
    int off = ch << 3;
    *(v8bf*)(ldsP + off) = *(const v8bf*)(projb + off);
  }
#endif
  // stage A scaled by data_normalizer
  const bf16_t* Abase = qk + ((size_t)bh * N_ + rowBase) * DH;
  for (int ch = t; ch < 512; ch += 128){
    int r = ch >> 3, cc = (ch & 7) << 3;
    v8bf v = *(const v8bf*)(Abase + r * DH + cc);
#pragma unroll
    for (int i = 0; i < 8; ++i) v[i] = f2b(b2f(v[i]) * dn);
    *(v8bf*)(ldsA + r * 64 + cc) = v;
  }
  __syncthreads();

  if (lane < 16){                               // diag = 0.5*||dn*q_row||^2
    float s = 0.f;
    const bf16_t* rp = ldsA + (wv * 16 + lane) * 64;
#pragma unroll 8
    for (int i = 0; i < 64; ++i){ float x = b2f(rp[i]); s += x * x; }
    diagBuf[wv][lane] = 0.5f * s;
  }
  __syncthreads();

  v8f acc[18];
#pragma unroll
  for (int c = 0; c < 18; ++c)
#pragma unroll
    for (int i = 0; i < 8; ++i) acc[c][i] = 0.f;

#pragma unroll
  for (int s = 0; s < 2; ++s){
    v16bf va = frag16(ldsA + wv * 16 * 64 + s * 32, 64);
#pragma unroll
    for (int tt = 0; tt < 18; ++tt){
      v16bf vb = frag16(ldsP + tt * 16 * 64 + s * 32, 64);
      acc[tt] = wmma_bf16(va, vb, acc[tt]);
    }
  }

  float rmax[8];
#pragma unroll
  for (int r = 0; r < 8; ++r) rmax[r] = -1e30f;
#pragma unroll
  for (int tt = 0; tt < 17; ++tt){
    bool valid = (tt < 16) || (ln < 10);        // exclude cols >= 266
    if (valid)
#pragma unroll
      for (int r = 0; r < 8; ++r) rmax[r] = fmaxf(rmax[r], acc[tt][r]);
  }
#pragma unroll
  for (int m = 8; m > 0; m >>= 1)
#pragma unroll
    for (int r = 0; r < 8; ++r)
      rmax[r] = fmaxf(rmax[r], __shfl_xor(rmax[r], m, 16));

  if (MODE == FEAT_K_STATS){
    if (ln == 0){
#pragma unroll
      for (int r = 0; r < 8; ++r)
        rowmax_out[(size_t)bh * N_ + rowBase + wv * 16 + r + 8 * half] = rmax[r];
    }
    return;
  }

  float stab_g = (MODE == FEAT_K_FINAL) ? stab_in[bh] : 0.f;
#pragma unroll
  for (int tt = 0; tt < 18; ++tt){
    int m = tt * 16 + ln;
#pragma unroll
    for (int r = 0; r < 8; ++r){
      int n = rowBase + wv * 16 + r + 8 * half;
      float diag = diagBuf[wv][r + 8 * half];
      float stab = (MODE == FEAT_Q) ? rmax[r] : stab_g;
      float val = (m < MF) ? ratio * (__expf(acc[tt][r] - diag - stab) + EPSF)
                           : 0.f;
      if (MODE == FEAT_Q)
        outQP[((size_t)bh * N_ + n) * MP + m] = f2b(val);
      else
        outKPT[((size_t)bh * MP + m) * N_ + n] = f2b(val);
    }
  }
}

__global__ __launch_bounds__(256)
void stab_reduce_kernel(const float* __restrict__ rowmax,
                        float* __restrict__ stab){
  __shared__ float red[256];
  int bh = blockIdx.x;
  float m = -1e30f;
  for (int i = threadIdx.x; i < N_; i += 256)
    m = fmaxf(m, rowmax[(size_t)bh * N_ + i]);
  red[threadIdx.x] = m;
  __syncthreads();
  for (int s = 128; s > 0; s >>= 1){
    if (threadIdx.x < s) red[threadIdx.x] = fmaxf(red[threadIdx.x], red[threadIdx.x + s]);
    __syncthreads();
  }
  if (threadIdx.x == 0) stab[bh] = red[0];
}

// --------------------- context = kp^T @ [v | 1 | 0-pad] --------------------
__global__ __launch_bounds__(64)
void ctx_kernel(const bf16_t* __restrict__ kpT,   // [BH][MP][N]
                const bf16_t* __restrict__ v,     // [BH][N][64]
                bf16_t* __restrict__ ctxT){       // [BH][80][MP]
  __shared__ bf16_t ldsA[32 * 32];
  __shared__ bf16_t ldsB[80 * 32];
  const int bh = blockIdx.y;
  const int mBase = blockIdx.x * 32;
  const int t = threadIdx.x;
  const int wv = t >> 5, lane = t & 31, half = lane >> 4, ln = lane & 15;

  v8f acc[5];
#pragma unroll
  for (int c = 0; c < 5; ++c)
#pragma unroll
    for (int i = 0; i < 8; ++i) acc[c][i] = 0.f;

  // constant augmented columns (c=64 -> ones, 65..79 -> zeros), set once
  const bf16_t one = f2b(1.0f);
  for (int idx = t; idx < 16 * 32; idx += 64){
    int c = 64 + (idx >> 5), kk = idx & 31;
    ldsB[c * 32 + kk] = (c == 64) ? one : (bf16_t)0.0f;
  }

  for (int k0 = 0; k0 < N_; k0 += 32){
    __syncthreads();
    // A tile 32x32 from kpT (row-major, contiguous)
#ifdef HAVE_TDM
    if (wv == 0){
      tdm_load_tile_2d(lds_off(ldsA),
                       kpT + ((size_t)bh * MP + mBase) * N_ + k0,
                       32, 32, N_, N_);
      __builtin_amdgcn_s_wait_tensorcnt(0);
    }
#else
#pragma unroll
    for (int i = 0; i < 2; ++i){
      int ch = t + 64 * i;
      int r = ch >> 2, cc = (ch & 3) << 3;
      *(v8bf*)(ldsA + r * 32 + cc) =
          *(const v8bf*)(kpT + ((size_t)bh * MP + mBase + r) * N_ + k0 + cc);
    }
#endif
    // B^T cols 0..63: vectorized row loads of v, transposed stores to LDS
    for (int idx = t; idx < 256; idx += 64){
      int kk = idx & 31, cg = idx >> 5;          // 32 rows x 8 col-groups
      v8bf vv = *(const v8bf*)(v + ((size_t)bh * N_ + k0 + kk) * DH + cg * 8);
#pragma unroll
      for (int j = 0; j < 8; ++j)
        ldsB[(cg * 8 + j) * 32 + kk] = vv[j];
    }
    __syncthreads();
    v16bf va = frag16(ldsA + wv * 16 * 32, 32);
#pragma unroll
    for (int c = 0; c < 5; ++c){
      v16bf vb = frag16(ldsB + c * 16 * 32, 32);
      acc[c] = wmma_bf16(va, vb, acc[c]);
    }
  }
#pragma unroll
  for (int c = 0; c < 5; ++c)
#pragma unroll
    for (int r = 0; r < 8; ++r){
      int gm = mBase + wv * 16 + r + 8 * half;
      int gc = c * 16 + ln;
      ctxT[((size_t)bh * 80 + gc) * MP + gm] = f2b(acc[c][r]);
    }
}

// ---------------- out = (qp @ ctx_aug) * (1 / (qp . k_sum)) ----------------
__global__ __launch_bounds__(128)
void attnout_kernel(const bf16_t* __restrict__ qp,    // [BH][N][MP]
                    const bf16_t* __restrict__ ctxT,  // [BH][80][MP]
                    bf16_t* __restrict__ attn){       // [BH][N][64]
  __shared__ bf16_t ldsA[64 * 32];
  __shared__ bf16_t ldsB[80 * 32];
  const int bh = blockIdx.y;
  const int rowBase = blockIdx.x * 64;
  const int t = threadIdx.x;
  const int wv = t >> 5, lane = t & 31, half = lane >> 4, ln = lane & 15;

  v8f acc[5];
#pragma unroll
  for (int c = 0; c < 5; ++c)
#pragma unroll
    for (int i = 0; i < 8; ++i) acc[c][i] = 0.f;

  for (int k0 = 0; k0 < MP; k0 += 32){
    __syncthreads();
#pragma unroll
    for (int i = 0; i < 2; ++i){              // A 64x32 from qp
      int ch = t + 128 * i;
      int r = ch >> 2, cc = (ch & 3) << 3;
      *(v8bf*)(ldsA + r * 32 + cc) =
          *(const v8bf*)(qp + ((size_t)bh * N_ + rowBase + r) * MP + k0 + cc);
    }
    // B^T 80x32 from ctxT (row-major)
#ifdef HAVE_TDM
    if (wv == 0){
      tdm_load_tile_2d(lds_off(ldsB),
                       ctxT + (size_t)bh * 80 * MP + k0,
                       32, 80, MP, MP);
      __builtin_amdgcn_s_wait_tensorcnt(0);
    }
#else
    for (int ch = t; ch < 320; ch += 128){
      int c = ch >> 2, cc = (ch & 3) << 3;
      *(v8bf*)(ldsB + c * 32 + cc) =
          *(const v8bf*)(ctxT + ((size_t)bh * 80 + c) * MP + k0 + cc);
    }
#endif
    __syncthreads();
    v16bf va = frag16(ldsA + wv * 16 * 32, 32);
#pragma unroll
    for (int c = 0; c < 5; ++c){
      v16bf vb = frag16(ldsB + c * 16 * 32, 32);
      acc[c] = wmma_bf16(va, vb, acc[c]);
    }
  }
#pragma unroll
  for (int r = 0; r < 8; ++r){
    float den = __shfl(acc[4][r], 0, 16);     // col 64 = qp . k_sum
    float inv = 1.0f / den;
    int n = rowBase + wv * 16 + r + 8 * half;
#pragma unroll
    for (int c = 0; c < 4; ++c){
      int d = c * 16 + ln;
      attn[((size_t)bh * N_ + n) * DH + d] = f2b(acc[c][r] * inv);
    }
  }
}

// ------------------------------- launcher ----------------------------------

extern "C" void kernel_launch(void* const* d_in, const int* in_sizes, int n_in,
                              void* d_out, int out_size, void* d_ws, size_t ws_size,
                              hipStream_t stream){
  const float* x    = (const float*)d_in[0];
  const float* Wq   = (const float*)d_in[1];
  const float* Wk   = (const float*)d_in[2];
  const float* Wv   = (const float*)d_in[3];
  const float* Wo   = (const float*)d_in[4];
  const float* bo   = (const float*)d_in[5];
  const float* proj = (const float*)d_in[6];
  float* out = (float*)d_out;

  char* p = (char*)d_ws;
  auto alloc = [&](size_t bytes) -> char* {
    char* r = p;
    p += (bytes + 255) & ~(size_t)255;
    return r;
  };
  const size_t ROWS = (size_t)B_ * N_;                  // 16384
  bf16_t* xb   = (bf16_t*)alloc(ROWS * DM * 2);
  bf16_t* wtq  = (bf16_t*)alloc((size_t)DM * DM * 2);
  bf16_t* wtk  = (bf16_t*)alloc((size_t)DM * DM * 2);
  bf16_t* wtv  = (bf16_t*)alloc((size_t)DM * DM * 2);
  bf16_t* wto  = (bf16_t*)alloc((size_t)DM * DM * 2);
  bf16_t* prjb = (bf16_t*)alloc((size_t)MP * DH * 2);
  bf16_t* qb   = (bf16_t*)alloc((size_t)BH * N_ * DH * 2);
  bf16_t* kb   = (bf16_t*)alloc((size_t)BH * N_ * DH * 2);
  bf16_t* vb   = (bf16_t*)alloc((size_t)BH * N_ * DH * 2);
  bf16_t* qp   = (bf16_t*)alloc((size_t)BH * N_ * MP * 2);
  bf16_t* kpT  = (bf16_t*)alloc((size_t)BH * MP * N_ * 2);
  bf16_t* ctxT = (bf16_t*)alloc((size_t)BH * 80 * MP * 2);
  bf16_t* attn = (bf16_t*)alloc((size_t)BH * N_ * DH * 2);
  float*  rmax = (float*)alloc((size_t)BH * N_ * 4);
  float*  stab = (float*)alloc((size_t)BH * 4);
  (void)ws_size; (void)n_in; (void)in_sizes; (void)out_size;

  // 0) precision conversions
  {
    int n = (int)(ROWS * DM);
    cvt_bf16_kernel<<<(n + 255) / 256, 256, 0, stream>>>(x, xb, n);
    transpose_cvt_1024<<<(DM * DM) / 256, 256, 0, stream>>>(Wq, wtq);
    transpose_cvt_1024<<<(DM * DM) / 256, 256, 0, stream>>>(Wk, wtk);
    transpose_cvt_1024<<<(DM * DM) / 256, 256, 0, stream>>>(Wv, wtv);
    transpose_cvt_1024<<<(DM * DM) / 256, 256, 0, stream>>>(Wo, wto);
    proj_pad_kernel<<<(MP * DH + 255) / 256, 256, 0, stream>>>(proj, prjb);
  }

  // 1) q/k/v projections: x @ W -> head layout bf16
  {
    dim3 grid(DM / 64, ROWS / 128);
    gemm1024_kernel<false, true><<<grid, 128, 0, stream>>>(xb, wtq, nullptr, qb, nullptr);
    gemm1024_kernel<false, true><<<grid, 128, 0, stream>>>(xb, wtk, nullptr, kb, nullptr);
    gemm1024_kernel<false, true><<<grid, 128, 0, stream>>>(xb, wtv, nullptr, vb, nullptr);
  }

  // 2) feature maps
  {
    dim3 grid(N_ / 64, BH);
    feat_kernel<FEAT_Q><<<grid, 128, 0, stream>>>(qb, prjb, qp, nullptr, nullptr, nullptr);
    feat_kernel<FEAT_K_STATS><<<grid, 128, 0, stream>>>(kb, prjb, nullptr, nullptr, rmax, nullptr);
    stab_reduce_kernel<<<BH, 256, 0, stream>>>(rmax, stab);
    feat_kernel<FEAT_K_FINAL><<<grid, 128, 0, stream>>>(kb, prjb, nullptr, kpT, nullptr, stab);
  }

  // 3) context (with fused k_sum column)
  {
    dim3 grid(MP / 32, BH);
    ctx_kernel<<<grid, 64, 0, stream>>>(kpT, vb, ctxT);
  }

  // 4) qp @ context, scaled by d_inv
  {
    dim3 grid(N_ / 64, BH);
    attnout_kernel<<<grid, 128, 0, stream>>>(qp, ctxT, attn);
  }

  // 5) final projection: attn(gathered) @ Wo + bo -> fp32 out
  {
    dim3 grid(DM / 64, ROWS / 128);
    gemm1024_kernel<true, false><<<grid, 128, 0, stream>>>(attn, wto, bo, nullptr, out);
  }
}
